// FailurePredictionGNN_12051678232960
// MI455X (gfx1250) — matrix operand order repaired and checked by reading, not compiled, not run on hardware
//
#include <hip/hip_runtime.h>
#include <math.h>

#define N_NODES 4096
#define E_EDGES 131072
#define DIN     128
#define HID     256
#define HEADS   8
#define DHD     32

typedef __attribute__((ext_vector_type(16))) _Float16 v16h;
typedef __attribute__((ext_vector_type(8)))  float    v8f;

// Per-lane K index of 16-bit pair j of an A fragment
// (ISA 7.12.2: lanes 0-15 hold K={0..7,16..23}, lanes 16-31 hold K={8..15,24..31})
__device__ __forceinline__ int kpairA(int j, int hi) {
    return (j < 4 ? 2 * j : 16 + 2 * (j - 4)) + (hi ? 8 : 0);
}
// Order used here for A and B: lanes 0-15 hold K=0..15 (contiguous pairs), lanes 16-31 hold K=16..31
__device__ __forceinline__ int kpairB(int j, int hi) {
    return 2 * j + (hi ? 16 : 0);
}

// LDS offset of a __shared__ object (generic ptr low 32 bits == LDS address)
__device__ __forceinline__ unsigned lds_off(const void* p) {
    return (unsigned)(size_t)p;
}
// Async DMA: 16 bytes global -> LDS, tracked by ASYNCcnt (CDNA5)
__device__ __forceinline__ void async_ld_b128(unsigned lds, const float* g) {
    asm volatile("global_load_async_to_lds_b128 %0, %1, off"
                 :: "v"(lds), "v"(g) : "memory");
}
__device__ __forceinline__ void wait_async0() {
    asm volatile("s_wait_asynccnt 0" ::: "memory");
}

// Max-reduce across each 16-lane row group using DPP row_ror ring steps
// (pure VALU; no LDS crossbar). row_ror:n ctrl = 0x120 | n.
__device__ __forceinline__ float rowmax16(float v) {
    int t;
    t = __builtin_amdgcn_update_dpp(0, __float_as_int(v), 0x121, 0xf, 0xf, true);
    v = fmaxf(v, __int_as_float(t));
    t = __builtin_amdgcn_update_dpp(0, __float_as_int(v), 0x122, 0xf, 0xf, true);
    v = fmaxf(v, __int_as_float(t));
    t = __builtin_amdgcn_update_dpp(0, __float_as_int(v), 0x124, 0xf, 0xf, true);
    v = fmaxf(v, __int_as_float(t));
    t = __builtin_amdgcn_update_dpp(0, __float_as_int(v), 0x128, 0xf, 0xf, true);
    v = fmaxf(v, __int_as_float(t));
    return v;
}

// ---------------------------------------------------------------------------
// WMMA GEMM:  C[M,Nn] = act( A[M,K] @ B (+ bias) ) (+ resid)
//   BT=false: B stored [K,Nn] row-major  (y = x @ W)      -> sB layout [k][col]
//   BT=true : B stored [Nn,K] row-major  (y = x @ W^T)    -> sB layout [col][k]
//   ACT: 0=none 1=relu 2=sigmoid
// Block = 128 threads (4 waves); block tile 64x16, one 16x16 tile per wave.
// Tiles staged as f32 via async-to-LDS b128 DMA; f16 convert at fragment gather.
// ---------------------------------------------------------------------------
template <int ACT, bool BT>
__global__ void __launch_bounds__(128)
gemm_wmma(const float* __restrict__ A, const float* __restrict__ B,
          const float* __restrict__ bias, const float* __restrict__ resid,
          float* __restrict__ C, int M, int Nn, int K) {
    __shared__ __align__(16) float sA[64 * 32];
    __shared__ __align__(16) float sB[512];         // [16][32] if BT else [32][16]
    const int tid  = threadIdx.x;
    const int wave = tid >> 5;
    const int lane = tid & 31;
    const int hi   = lane >> 4;
    const int l16  = lane & 15;
    const int m0   = blockIdx.y * 64;
    const int n0   = blockIdx.x * 16;

    v8f acc = {};
    for (int k0 = 0; k0 < K; k0 += 32) {
        __syncthreads();   // previous tile fully consumed before restaging
        // ---- stage A 64x32 via async DMA: 512 float4, 4 per thread ----
#pragma unroll
        for (int q = 0; q < 4; ++q) {
            int idx4 = q * 128 + tid;               // 0..511
            int r = idx4 >> 3, c4 = (idx4 & 7) * 4;
            async_ld_b128(lds_off(&sA[r * 32 + c4]),
                          A + (size_t)(m0 + r) * K + k0 + c4);
        }
        // ---- stage B 32x16 ----
        if (BT) {
            if (n0 + 16 <= Nn) {                    // full tile: DMA, [col][k]
                int col = tid >> 3, kq = (tid & 7) * 4;
                async_ld_b128(lds_off(&sB[col * 32 + kq]),
                              B + (size_t)(n0 + col) * K + k0 + kq);
            } else {                                // ragged (N=3/10): guarded
#pragma unroll
                for (int i = 0; i < 4; ++i) {
                    int idx = tid * 4 + i;          // 0..511
                    int col = idx >> 5, kk = idx & 31;
                    sB[idx] = (n0 + col < Nn)
                                  ? B[(size_t)(n0 + col) * K + k0 + kk] : 0.f;
                }
            }
        } else {                                    // [k][col], contiguous in n
            int kk = tid >> 2, c4 = (tid & 3) * 4;
            async_ld_b128(lds_off(&sB[kk * 16 + c4]),
                          B + (size_t)(k0 + kk) * Nn + n0 + c4);
        }
        wait_async0();
        __syncthreads();

        // ---- gather fragments (f32 LDS -> f16 regs) ----
        v16h af, bf;
        const int arow = wave * 16 + l16;
        const float2* ar0 = (const float2*)(sA + arow * 32 + hi * 16);
        const float2* ar1 = (const float2*)(sA + arow * 32 + hi * 16 + 8);
#pragma unroll
        for (int j = 0; j < 4; ++j) {
            float2 p = ar0[j];
            af[2 * j]     = (_Float16)p.x;
            af[2 * j + 1] = (_Float16)p.y;
            float2 q = ar1[j];
            af[2 * (j + 4)]     = (_Float16)q.x;
            af[2 * (j + 4) + 1] = (_Float16)q.y;
        }
        if (BT) {
            const float2* br = (const float2*)(sB + l16 * 32 + hi * 16);
#pragma unroll
            for (int j = 0; j < 8; ++j) {
                float2 p = br[j];
                bf[2 * j]     = (_Float16)p.x;
                bf[2 * j + 1] = (_Float16)p.y;
            }
        } else {
#pragma unroll
            for (int j = 0; j < 8; ++j) {
                int kb = kpairB(j, hi);
                bf[2 * j]     = (_Float16)sB[kb * 16 + l16];
                bf[2 * j + 1] = (_Float16)sB[(kb + 1) * 16 + l16];
            }
        }
        acc = __builtin_amdgcn_wmma_f32_16x16x32_f16(false, af, false, bf,
                                                     (short)0, acc, false, false);
    }

    // Epilogue: C layout -> row = j + hi*8, col = l16
#pragma unroll
    for (int j = 0; j < 8; ++j) {
        int r = m0 + wave * 16 + j + hi * 8;
        int n = n0 + l16;
        if (n < Nn) {
            float v = acc[j];
            if (bias) v += bias[n];
            if (ACT == 1) v = v > 0.f ? v : 0.f;
            else if (ACT == 2) v = 1.f / (1.f + __expf(-v));
            if (resid) v += resid[(size_t)r * Nn + n];
            C[(size_t)r * Nn + n] = v;
        }
    }
}

// ---------------------------------------------------------------------------
// Fused flash attention: qkv[N, 3H] -> out[N, H]
// grid(N/64, HEADS), block 128. All 4 waves share one head: K/V tiles
// (32 keys x 32 dh) staged once per block via async DMA. Row-sum of P is
// computed by a WMMA against an all-ones B fragment; row-max via DPP ring.
// ---------------------------------------------------------------------------
__global__ void __launch_bounds__(128)
flash_attn(const float* __restrict__ qkv, float* __restrict__ out) {
    __shared__ __align__(16) float sK[32 * 32];
    __shared__ __align__(16) float sV[32 * 32];
    __shared__ _Float16 sP[4][16 * 32];     // wave-private P staging
    const int tid  = threadIdx.x;
    const int wave = tid >> 5;
    const int lane = tid & 31;
    const int hi   = lane >> 4;
    const int l16  = lane & 15;
    const int m0   = blockIdx.x * 64 + wave * 16;
    const int hd   = blockIdx.y;
    const int RS   = 3 * HID;
    const int qo   = hd * DHD;
    const int ko   = HID + hd * DHD;
    const int vo   = 2 * HID + hd * DHD;
    const float scale = 0.17677669529663687f;   // 1/sqrt(32)

    // Q fragment (A layout), scale folded in; two contiguous 8-float spans
    v16h qf;
    {
        const float4* q4 = (const float4*)(qkv + (size_t)(m0 + l16) * RS + qo);
        float4 a = q4[4 * hi],     b = q4[4 * hi + 1];      // d = hi*16 + 0..7
        float4 c = q4[4 * hi + 2], d = q4[4 * hi + 3];      // d = hi*16+8+0..7
        qf[0] = (_Float16)(a.x * scale);  qf[1] = (_Float16)(a.y * scale);
        qf[2] = (_Float16)(a.z * scale);  qf[3] = (_Float16)(a.w * scale);
        qf[4] = (_Float16)(b.x * scale);  qf[5] = (_Float16)(b.y * scale);
        qf[6] = (_Float16)(b.z * scale);  qf[7] = (_Float16)(b.w * scale);
        qf[8]  = (_Float16)(c.x * scale); qf[9]  = (_Float16)(c.y * scale);
        qf[10] = (_Float16)(c.z * scale); qf[11] = (_Float16)(c.w * scale);
        qf[12] = (_Float16)(d.x * scale); qf[13] = (_Float16)(d.y * scale);
        qf[14] = (_Float16)(d.z * scale); qf[15] = (_Float16)(d.w * scale);
    }
    v16h ones;
#pragma unroll
    for (int j = 0; j < 16; ++j) ones[j] = (_Float16)1.0f;

    float mrow[8];
#pragma unroll
    for (int j = 0; j < 8; ++j) mrow[j] = -1e30f;
    v8f O0 = {}, O1 = {}, L = {};
    _Float16* sp = &sP[wave][0];

    for (int jb = 0; jb < N_NODES; jb += 32) {
        __syncthreads();   // previous K/V tile fully consumed
        // Stage K,V tiles (32x32 f32 each) via async DMA: 2+2 b128 per thread
#pragma unroll
        for (int q = 0; q < 2; ++q) {
            int idx4 = q * 128 + tid;               // 0..255
            int r = idx4 >> 3, c4 = (idx4 & 7) * 4;
            const float* g = qkv + (size_t)(jb + r) * RS;
            async_ld_b128(lds_off(&sK[r * 32 + c4]), g + ko + c4);
            async_ld_b128(lds_off(&sV[r * 32 + c4]), g + vo + c4);
        }
        wait_async0();
        __syncthreads();

        // K^T fragments: B[d][col] = K[col][d]; contiguous d pairs per lane
        v16h kb0, kb1;
        {
            const float2* k0p = (const float2*)(sK + l16 * 32 + hi * 16);
            const float2* k1p = (const float2*)(sK + (16 + l16) * 32 + hi * 16);
#pragma unroll
            for (int j = 0; j < 8; ++j) {
                float2 a = k0p[j], b = k1p[j];
                kb0[2 * j] = (_Float16)a.x;  kb0[2 * j + 1] = (_Float16)a.y;
                kb1[2 * j] = (_Float16)b.x;  kb1[2 * j + 1] = (_Float16)b.y;
            }
        }
        v8f z = {};
        v8f s0 = __builtin_amdgcn_wmma_f32_16x16x32_f16(false, qf, false, kb0,
                                                        (short)0, z, false, false);
        v8f s1 = __builtin_amdgcn_wmma_f32_16x16x32_f16(false, qf, false, kb1,
                                                        (short)0, z, false, false);
        // Online softmax: row max via DPP ring reduction (no LDS traffic)
#pragma unroll
        for (int j = 0; j < 8; ++j) {
            float mx = rowmax16(fmaxf(s0[j], s1[j]));
            float mnew = fmaxf(mrow[j], mx);
            float corr = __expf(mrow[j] - mnew);
            mrow[j] = mnew;
            float p0 = __expf(s0[j] - mnew);
            float p1 = __expf(s1[j] - mnew);
            O0[j] *= corr;
            O1[j] *= corr;
            L[j]  *= corr;
            int r = j + hi * 8;                // C-layout row
            sp[r * 32 + l16]      = (_Float16)p0;
            sp[r * 32 + 16 + l16] = (_Float16)p1;
        }
        // P as A fragment (via wave-private LDS transpose), V as B fragments
        v16h pa, vb0, vb1;
#pragma unroll
        for (int j = 0; j < 8; ++j) {
            int d = kpairB(j, hi);
            pa[2 * j]     = sp[l16 * 32 + d];
            pa[2 * j + 1] = sp[l16 * 32 + d + 1];
            int kk = kpairB(j, hi);
            vb0[2 * j]     = (_Float16)sV[kk * 32 + l16];
            vb0[2 * j + 1] = (_Float16)sV[(kk + 1) * 32 + l16];
            vb1[2 * j]     = (_Float16)sV[kk * 32 + 16 + l16];
            vb1[2 * j + 1] = (_Float16)sV[(kk + 1) * 32 + 16 + l16];
        }
        L  = __builtin_amdgcn_wmma_f32_16x16x32_f16(false, pa, false, ones,
                                                    (short)0, L, false, false);
        O0 = __builtin_amdgcn_wmma_f32_16x16x32_f16(false, pa, false, vb0,
                                                    (short)0, O0, false, false);
        O1 = __builtin_amdgcn_wmma_f32_16x16x32_f16(false, pa, false, vb1,
                                                    (short)0, O1, false, false);
    }
#pragma unroll
    for (int j = 0; j < 8; ++j) {
        int r = m0 + j + hi * 8;
        float inv = 1.f / L[j];
        out[(size_t)r * HID + hd * DHD + l16]      = O0[j] * inv;
        out[(size_t)r * HID + hd * DHD + 16 + l16] = O1[j] * inv;
    }
}

// ---------------------------------------------------------------------------
// Scalar helper kernels
// ---------------------------------------------------------------------------
__global__ void k_fill(float* p, float v, int n) {
    int i = blockIdx.x * blockDim.x + threadIdx.x;
    if (i < n) p[i] = v;
}
__global__ void k_deg_count(const int* __restrict__ dst, float* __restrict__ deg) {
    int e = blockIdx.x * blockDim.x + threadIdx.x;
    if (e < E_EDGES) atomicAdd(&deg[dst[e]], 1.f);
}
__global__ void k_dinv(const float* __restrict__ deg, float* __restrict__ dinv,
                       float* __restrict__ invdeg) {
    int i = blockIdx.x * blockDim.x + threadIdx.x;
    if (i < N_NODES) {
        float d = deg[i];
        dinv[i] = rsqrtf(d);
        invdeg[i] = 1.f / d;
    }
}
__global__ void k_agg_init(float* __restrict__ out, const float* __restrict__ h,
                           const float* __restrict__ invdeg,
                           const float* __restrict__ bias) {
    int idx = blockIdx.x * blockDim.x + threadIdx.x;
    if (idx < N_NODES * HID) {
        int n = idx >> 8, c = idx & 255;   // HID == 256
        out[idx] = h[idx] * invdeg[n] + bias[c];
    }
}
__global__ void k_scatter(float* __restrict__ out, const float* __restrict__ h,
                          const int* __restrict__ src, const int* __restrict__ dst,
                          const float* __restrict__ dinv) {
    int idx = blockIdx.x * blockDim.x + threadIdx.x;
    if (idx >= E_EDGES * 8) return;
    int e = idx >> 3, ch = (idx & 7) * 32;
    int s = src[e], d = dst[e];
    float coef = dinv[s] * dinv[d];
    const float* hp = h + (size_t)s * HID + ch;
    float* op = out + (size_t)d * HID + ch;
#pragma unroll
    for (int c = 0; c < 32; ++c) atomicAdd(&op[c], hp[c] * coef);
}
__global__ void k_relu(float* p, int n) {
    int i = blockIdx.x * blockDim.x + threadIdx.x;
    if (i < n) p[i] = fmaxf(p[i], 0.f);
}
__global__ void k_softmax3(const float* __restrict__ logits, float* __restrict__ out) {
    int n = blockIdx.x * blockDim.x + threadIdx.x;
    if (n < N_NODES) {
        float a = logits[n * 3], b = logits[n * 3 + 1], c = logits[n * 3 + 2];
        float m = fmaxf(a, fmaxf(b, c));
        float ea = __expf(a - m), eb = __expf(b - m), ec = __expf(c - m);
        float s = ea + eb + ec;
        out[n * 3]     = ea / s;
        out[n * 3 + 1] = eb / s;
        out[n * 3 + 2] = ec / s;
    }
}

// ---------------------------------------------------------------------------
extern "C" void kernel_launch(void* const* d_in, const int* in_sizes, int n_in,
                              void* d_out, int out_size, void* d_ws, size_t ws_size,
                              hipStream_t stream) {
    const float* x     = (const float*)d_in[0];
    const int*   ei    = (const int*)d_in[1];
    const float* W1    = (const float*)d_in[2];
    const float* b1    = (const float*)d_in[3];
    const float* W2    = (const float*)d_in[4];
    const float* b2    = (const float*)d_in[5];
    const float* W3    = (const float*)d_in[6];
    const float* b3    = (const float*)d_in[7];
    const float* in_w  = (const float*)d_in[8];
    const float* in_b  = (const float*)d_in[9];
    const float* out_w = (const float*)d_in[10];
    const float* out_b = (const float*)d_in[11];
    const float* fp1_w = (const float*)d_in[12];
    const float* fp1_b = (const float*)d_in[13];
    const float* fp2_w = (const float*)d_in[14];
    const float* fp2_b = (const float*)d_in[15];
    const float* pd1_w = (const float*)d_in[16];
    const float* pd1_b = (const float*)d_in[17];
    const float* pd2_w = (const float*)d_in[18];
    const float* pd2_b = (const float*)d_in[19];
    const int* srcI = ei;
    const int* dstI = ei + E_EDGES;

    float* ws     = (float*)d_ws;
    float* deg    = ws;                       //   4096
    float* dinv   = ws + 4096;
    float* invdeg = ws + 8192;
    float* tmp    = ws + 16384;               // 1M
    float* bufA   = tmp + (1 << 20);          // 1M
    float* bufB   = bufA + (1 << 20);         // 1M
    float* qkvb   = bufB + (1 << 20);         // 3M
    float* attnO  = qkvb + 3 * (1 << 20);     // 1M  (total ~29.4 MB)
    float* f1     = tmp;                      // reuse for MLP hidden [4096,128]
    float* logit3 = tmp + N_NODES * DIN;      // fp2 logits [4096,3]
    float* risk   = (float*)d_out;
    float* patt   = (float*)d_out + N_NODES * 3;

    const int EW = 256;
    dim3 gNH((N_NODES * HID + EW - 1) / EW), bEW(EW);
    dim3 gN((N_NODES + EW - 1) / EW);
    dim3 gE((E_EDGES + EW - 1) / EW);
    dim3 gE8((E_EDGES * 8 + EW - 1) / EW);
    dim3 gmB(128);

    k_fill<<<gN, bEW, 0, stream>>>(deg, 1.f, N_NODES);
    k_deg_count<<<gE, bEW, 0, stream>>>(dstI, deg);
    k_dinv<<<gN, bEW, 0, stream>>>(deg, dinv, invdeg);

    // GCN layer 1
    gemm_wmma<0, false><<<dim3(HID / 16, N_NODES / 64), gmB, 0, stream>>>(
        x, W1, nullptr, nullptr, tmp, N_NODES, HID, DIN);
    k_agg_init<<<gNH, bEW, 0, stream>>>(bufA, tmp, invdeg, b1);
    k_scatter<<<gE8, bEW, 0, stream>>>(bufA, tmp, srcI, dstI, dinv);
    k_relu<<<gNH, bEW, 0, stream>>>(bufA, N_NODES * HID);

    // GCN layer 2
    gemm_wmma<0, false><<<dim3(HID / 16, N_NODES / 64), gmB, 0, stream>>>(
        bufA, W2, nullptr, nullptr, tmp, N_NODES, HID, HID);
    k_agg_init<<<gNH, bEW, 0, stream>>>(bufB, tmp, invdeg, b2);
    k_scatter<<<gE8, bEW, 0, stream>>>(bufB, tmp, srcI, dstI, dinv);
    k_relu<<<gNH, bEW, 0, stream>>>(bufB, N_NODES * HID);

    // GCN layer 3 (no relu) -> bufA = h3
    gemm_wmma<0, false><<<dim3(HID / 16, N_NODES / 64), gmB, 0, stream>>>(
        bufB, W3, nullptr, nullptr, tmp, N_NODES, HID, HID);
    k_agg_init<<<gNH, bEW, 0, stream>>>(bufA, tmp, invdeg, b3);
    k_scatter<<<gE8, bEW, 0, stream>>>(bufA, tmp, srcI, dstI, dinv);

    // MHA
    gemm_wmma<0, true><<<dim3(3 * HID / 16, N_NODES / 64), gmB, 0, stream>>>(
        bufA, in_w, in_b, nullptr, qkvb, N_NODES, 3 * HID, HID);
    flash_attn<<<dim3(N_NODES / 64, HEADS), gmB, 0, stream>>>(qkvb, attnO);
    gemm_wmma<0, true><<<dim3(HID / 16, N_NODES / 64), gmB, 0, stream>>>(
        attnO, out_w, out_b, bufA, bufB, N_NODES, HID, HID);

    // failure_predictor
    gemm_wmma<1, true><<<dim3(DIN / 16, N_NODES / 64), gmB, 0, stream>>>(
        bufB, fp1_w, fp1_b, nullptr, f1, N_NODES, DIN, HID);
    gemm_wmma<0, true><<<dim3(1, N_NODES / 64), gmB, 0, stream>>>(
        f1, fp2_w, fp2_b, nullptr, logit3, N_NODES, 3, DIN);
    k_softmax3<<<gN, bEW, 0, stream>>>(logit3, risk);

    // pattern_detector
    gemm_wmma<1, true><<<dim3(DIN / 16, N_NODES / 64), gmB, 0, stream>>>(
        bufB, pd1_w, pd1_b, nullptr, f1, N_NODES, DIN, HID);
    gemm_wmma<2, true><<<dim3(1, N_NODES / 64), gmB, 0, stream>>>(
        f1, pd2_w, pd2_b, nullptr, patt, N_NODES, 10, DIN);
}
